// PointNetSetAbstraction_3186865734537
// MI455X (gfx1250) — matrix operand order repaired and checked by reading, not compiled, not run on hardware
//
#include <hip/hip_runtime.h>
#include <hip/hip_bf16.h>

// ---------------- problem constants ----------------
#define BATCH   16
#define NPTS    4096
#define NPOINT  1024
#define NSAMPLE 64
#define RAD2    0.04f            // 0.2^2
#define M_ROWS  (BATCH * NPOINT * NSAMPLE)   // 1,048,576
#define EPSBN   1e-5f

typedef __attribute__((ext_vector_type(16))) __bf16 v16bf;
typedef __attribute__((ext_vector_type(8)))  __bf16 v8bf;
typedef __attribute__((ext_vector_type(8)))  float  v8f;
typedef __attribute__((ext_vector_type(4)))  int    v4i;

// ---------------- workspace layout (bytes) ----------------
#define O_SUMS   ((size_t)0)            // 3 layers * 256 f32 (sum[128], sumsq[128])
#define O_SCALE  ((size_t)3072)         // 3 layers * 256 f32 (scale[128], shift[128])
#define O_FPS    ((size_t)8192)         // B*NPOINT int
#define O_WB0    ((size_t)(8192 + 65536))           // 64*32 bf16
#define O_WB1    (O_WB0 + 4096)                     // 64*64 bf16
#define O_WB2    (O_WB1 + 8192)                     // 128*64 bf16
#define O_X0     ((size_t)131072)                   // M*32 bf16 = 64 MiB
#define O_XB     (O_X0 + (size_t)M_ROWS * 32 * 2)   // M*64 bf16 = 128 MiB
#define O_Y3     (O_XB + (size_t)M_ROWS * 64 * 2)   // M*128 bf16 = 256 MiB

// ================= prep: weights f32->bf16 (zero-padded), zero BN sums =================
__global__ void prep_kernel(const float* __restrict__ w0, const float* __restrict__ w1,
                            const float* __restrict__ w2,
                            __hip_bfloat16* wb0, __hip_bfloat16* wb1, __hip_bfloat16* wb2,
                            float* sums) {
  int i = blockIdx.x * blockDim.x + threadIdx.x;
  int stride = gridDim.x * blockDim.x;
  for (int t = i; t < 64 * 32; t += stride) {               // w0: (64,9) -> (64,32) padded
    int o = t >> 5, k = t & 31;
    wb0[t] = __float2bfloat16(k < 9 ? w0[o * 9 + k] : 0.0f);
  }
  for (int t = i; t < 64 * 64; t += stride) wb1[t] = __float2bfloat16(w1[t]);
  for (int t = i; t < 128 * 64; t += stride) wb2[t] = __float2bfloat16(w2[t]);
  for (int t = i; t < 3 * 256; t += stride) sums[t] = 0.0f;
}

// ================= FPS: one 1024-thread block per batch =================
__global__ void __launch_bounds__(1024) fps_kernel(const float* __restrict__ xyz,
                                                   int* __restrict__ fpsIdx,
                                                   float* __restrict__ newXyz) {
  const int b = blockIdx.x;
  const int tid = threadIdx.x;                  // 0..1023
  const float* xb = xyz + (size_t)b * NPTS * 3;
  float px[4], py[4], pz[4], dist[4];
#pragma unroll
  for (int i = 0; i < 4; ++i) {
    int p = tid + i * 1024;
    px[i] = xb[p * 3 + 0]; py[i] = xb[p * 3 + 1]; pz[i] = xb[p * 3 + 2];
    dist[i] = 1e10f;
  }
  __shared__ float sVal[32];
  __shared__ int   sIdx[32];
  __shared__ float sC[3];
  __shared__ int   sFar;
  int far = 0;
  for (int it = 0; it < NPOINT; ++it) {
    if (tid == 0) {
      fpsIdx[b * NPOINT + it] = far;
      float cx = xb[far * 3 + 0], cy = xb[far * 3 + 1], cz = xb[far * 3 + 2];
      sC[0] = cx; sC[1] = cy; sC[2] = cz;
      float* o = newXyz + ((size_t)b * NPOINT + it) * 3;
      o[0] = cx; o[1] = cy; o[2] = cz;
    }
    __syncthreads();
    float cx = sC[0], cy = sC[1], cz = sC[2];
    float best = -1.0f; int bi = tid;
#pragma unroll
    for (int i = 0; i < 4; ++i) {
      float dx = px[i] - cx, dy = py[i] - cy, dz = pz[i] - cz;
      float d = dx * dx + dy * dy + dz * dz;
      dist[i] = fminf(dist[i], d);
      if (dist[i] > best) { best = dist[i]; bi = tid + i * 1024; }
    }
#pragma unroll
    for (int off = 16; off > 0; off >>= 1) {    // wave32 reduce
      float ov = __shfl_xor(best, off, 32);
      int   oi = __shfl_xor(bi,   off, 32);
      if (ov > best || (ov == best && oi < bi)) { best = ov; bi = oi; }
    }
    if ((tid & 31) == 0) { sVal[tid >> 5] = best; sIdx[tid >> 5] = bi; }
    __syncthreads();
    if (tid < 32) {
      best = sVal[tid]; bi = sIdx[tid];
#pragma unroll
      for (int off = 16; off > 0; off >>= 1) {
        float ov = __shfl_xor(best, off, 32);
        int   oi = __shfl_xor(bi,   off, 32);
        if (ov > best || (ov == best && oi < bi)) { best = ov; bi = oi; }
      }
      if (tid == 0) sFar = bi;
    }
    __syncthreads();
    far = sFar;
  }
}

// ================= ball query + group + concat: one wave per centroid =================
__global__ void __launch_bounds__(256) ball_group_kernel(const float* __restrict__ xyz,
                                                         const float* __restrict__ pts,
                                                         const float* __restrict__ newXyz,
                                                         __hip_bfloat16* __restrict__ X0) {
  const int lane = threadIdx.x & 31;
  const int wib  = threadIdx.x >> 5;
  const int cent = blockIdx.x * 8 + wib;        // 0..16383
  const int b = cent >> 10;
  const float* xb = xyz + (size_t)b * NPTS * 3;
  const float* pb = pts + (size_t)b * NPTS * 6;
  const float* c  = newXyz + (size_t)cent * 3;
  const float cx = c[0], cy = c[1], cz = c[2];

  __shared__ int sSel[8][NSAMPLE];
  int count = 0;
  for (int n0 = 0; n0 < NPTS && count < NSAMPLE; n0 += 32) {
    int p = n0 + lane;
    float dx = xb[p * 3 + 0] - cx, dy = xb[p * 3 + 1] - cy, dz = xb[p * 3 + 2] - cz;
    float d = dx * dx + dy * dy + dz * dz;
    bool pred = (d <= RAD2);                    // reference excludes sqr > r^2
    unsigned mask = (unsigned)__ballot(pred);
    int pos = count + __popc(mask & ((1u << lane) - 1u));
    if (pred && pos < NSAMPLE) sSel[wib][pos] = p;
    count += __popc(mask);
  }
  int total = count < NSAMPLE ? count : NSAMPLE;
  if (total == 0) { if (lane == 0) sSel[wib][0] = 0; total = 1; }

  for (int k = lane; k < NSAMPLE; k += 32) {
    int idx = (k < total) ? sSel[wib][k] : sSel[wib][0];
    const float* q = xb + idx * 3;
    const float* f = pb + (size_t)idx * 6;
    union { __hip_bfloat16 h[32]; uint4 q4[4]; } row;
    row.h[0] = __float2bfloat16(q[0] - cx);
    row.h[1] = __float2bfloat16(q[1] - cy);
    row.h[2] = __float2bfloat16(q[2] - cz);
#pragma unroll
    for (int j = 0; j < 6; ++j) row.h[3 + j] = __float2bfloat16(f[j]);
#pragma unroll
    for (int j = 9; j < 32; ++j) row.h[j] = __float2bfloat16(0.0f);
    uint4* dst = (uint4*)(X0 + ((size_t)cent * NSAMPLE + k) * 32);
    dst[0] = row.q4[0]; dst[1] = row.q4[1]; dst[2] = row.q4[2]; dst[3] = row.q4[3];
  }
}

// ================= bf16 WMMA GEMM: preact = X(M x KPAD) * W^T + bias =================
// Block: 64-row A slab in LDS; one wave per 16-col N tile; each wave keeps its B
// fragments in registers and sweeps 4 M-tiles (B reuse, 4-8 WMMA per wave).
template <int KPAD, int COUT>
__global__ void __launch_bounds__(256) gemm_kernel(const __hip_bfloat16* __restrict__ X,
                                                   const __hip_bfloat16* __restrict__ Wb,
                                                   const float* __restrict__ bias,
                                                   __hip_bfloat16* __restrict__ Y) {
  __shared__ __align__(32) __hip_bfloat16 ldsA[64 * KPAD];
  const int mBase = blockIdx.x * 64;
  const int tid = threadIdx.x;
  constexpr int NTHR = (COUT / 16) * 32;
  constexpr int chunks = 64 * KPAD / 8;          // 16B (v4i) chunks
  const __hip_bfloat16* gsrc = X + (size_t)mBase * KPAD;

#if defined(__HIP_DEVICE_COMPILE__) && \
    __has_builtin(__builtin_amdgcn_global_load_async_to_lds_b128)
  {
    v4i* lbase = (v4i*)&ldsA[0];                 // generic ptr -> implicit cast to LDS AS
    v4i* gbase = (v4i*)gsrc;                     // generic ptr -> implicit cast to global AS
    for (int t = tid; t < chunks; t += NTHR)
      __builtin_amdgcn_global_load_async_to_lds_b128(gbase + t, lbase + t, 0, 0);
#if __has_builtin(__builtin_amdgcn_s_wait_asynccnt)
    __builtin_amdgcn_s_wait_asynccnt(0);
#else
    asm volatile("s_wait_asynccnt 0x0" ::: "memory");
#endif
  }
#else
  for (int t = tid; t < chunks; t += NTHR)
    ((uint4*)ldsA)[t] = ((const uint4*)gsrc)[t];
#endif
  __syncthreads();

  const int lane = tid & 31;
  const int n    = ((tid >> 5) << 4) + (lane & 15);   // output channel
  const int kg   = lane >> 4;                          // 0/1 half-wave K group

  v16bf Bv[KPAD / 32];                                 // B fragments, register-resident
#pragma unroll
  for (int k0 = 0; k0 < KPAD; k0 += 32) {
    union { v16bf v; v8bf h[2]; } Bu;
    Bu.h[0] = *(const v8bf*)(Wb + (size_t)n * KPAD + k0 + kg * 8);
    Bu.h[1] = *(const v8bf*)(Wb + (size_t)n * KPAD + k0 + 16 + kg * 8);
    Bv[k0 / 32] = Bu.v;
  }
  const float bb = bias[n];

#pragma unroll
  for (int mt = 0; mt < 4; ++mt) {
    v8f acc = {};
#pragma unroll
    for (int k0 = 0; k0 < KPAD; k0 += 32) {
      union { v16bf v; v8bf h[2]; } A;
      const __hip_bfloat16* ar = ldsA + (mt * 16 + (lane & 15)) * KPAD + k0;
      A.h[0] = *(const v8bf*)(ar + kg * 8);
      A.h[1] = *(const v8bf*)(ar + 16 + kg * 8);
      acc = __builtin_amdgcn_wmma_f32_16x16x32_bf16(false, A.v, false, Bv[k0 / 32],
                                                    (short)0, acc, false, false);
    }
#pragma unroll
    for (int r = 0; r < 8; ++r) {
      int m = mBase + mt * 16 + r + (kg << 3);        // C layout: rows 0-7 / 8-15
      Y[(size_t)m * COUT + n] = __float2bfloat16(acc[r] + bb);
    }
  }
}

// ================= per-channel sum / sumsq over M rows (uint4-vectorized) =================
template <int COUT>
__global__ void __launch_bounds__(256) stats_kernel(const __hip_bfloat16* __restrict__ Y,
                                                    float* __restrict__ sums) {
  constexpr int VPR = COUT / 8;                 // uint4 vectors per row
  const int tid = threadIdx.x;
  const size_t nv = (size_t)M_ROWS * COUT / 8;
  const size_t i0 = (size_t)blockIdx.x * 256 + tid;
  const size_t stride = (size_t)gridDim.x * 256;   // (stride*8) % COUT == 0
  float s[8], s2[8];
#pragma unroll
  for (int j = 0; j < 8; ++j) { s[j] = 0.0f; s2[j] = 0.0f; }
  for (size_t i = i0; i < nv; i += stride) {
    uint4 v = ((const uint4*)Y)[i];
    const __hip_bfloat16* h = (const __hip_bfloat16*)&v;
#pragma unroll
    for (int j = 0; j < 8; ++j) {
      float f = __bfloat162float(h[j]);
      s[j] += f; s2[j] += f * f;
    }
  }
  __shared__ float ls[256][8], ls2[256][8];
#pragma unroll
  for (int j = 0; j < 8; ++j) { ls[tid][j] = s[j]; ls2[tid][j] = s2[j]; }
  __syncthreads();
  if (tid < COUT) {
    const int tb = tid >> 3, j = tid & 7;
    float a = 0.0f, a2 = 0.0f;
#pragma unroll
    for (int m = 0; m < 256 / VPR; ++m) { a += ls[tb + m * VPR][j]; a2 += ls2[tb + m * VPR][j]; }
    atomicAdd(&sums[tid], a);
    atomicAdd(&sums[128 + tid], a2);
  }
}

// ================= BN finalize: scale/shift per channel =================
__global__ void finalize_kernel(const float* __restrict__ sums, const float* __restrict__ g,
                                const float* __restrict__ be, float* __restrict__ sc, int cout) {
  int c = threadIdx.x;
  if (c < cout) {
    const float invM = 1.0f / (float)M_ROWS;
    float mean = sums[c] * invM;
    float var  = sums[128 + c] * invM - mean * mean;
    float s    = g[c] * rsqrtf(var + EPSBN);
    sc[c]       = s;
    sc[128 + c] = be[c] - mean * s;
  }
}

// ================= BN + ReLU elementwise (in place, uint4-vectorized) =================
template <int COUT>
__global__ void __launch_bounds__(256) bnrelu_kernel(__hip_bfloat16* __restrict__ Y,
                                                     const float* __restrict__ sc) {
  const size_t nv = (size_t)M_ROWS * COUT / 8;
  const size_t i0 = (size_t)blockIdx.x * 256 + threadIdx.x;
  const size_t stride = (size_t)gridDim.x * 256;
  const int c0 = (int)((i0 * 8) & (COUT - 1));
  for (size_t i = i0; i < nv; i += stride) {
    uint4 v = ((const uint4*)Y)[i];
    __hip_bfloat16* h = (__hip_bfloat16*)&v;
#pragma unroll
    for (int j = 0; j < 8; ++j) {
      float f = __bfloat162float(h[j]);
      f = fmaxf(sc[c0 + j] * f + sc[128 + c0 + j], 0.0f);
      h[j] = __float2bfloat16(f);
    }
    ((uint4*)Y)[i] = v;
  }
}

// ================= layer-3 BN+ReLU fused with max over NSAMPLE + transpose =================
__global__ void __launch_bounds__(256) maxpool_kernel(const __hip_bfloat16* __restrict__ Y3,
                                                      const float* __restrict__ sc,
                                                      float* __restrict__ out) {
  int gid = blockIdx.x * 256 + threadIdx.x;       // (b, s, c) over 16*1024*128
  int c = gid & 127;
  int s = (gid >> 7) & 1023;
  int b = gid >> 17;
  float scale = sc[c], shift = sc[128 + c];
  const __hip_bfloat16* p = Y3 + ((size_t)(b * NPOINT + s) * NSAMPLE) * 128 + c;
  float m = 0.0f;                                  // relu -> max >= 0
#pragma unroll 4
  for (int k = 0; k < NSAMPLE; ++k) {
    float v = fmaxf(scale * __bfloat162float(p[(size_t)k * 128]) + shift, 0.0f);
    m = fmaxf(m, v);
  }
  out[((size_t)b * 128 + c) * NPOINT + s] = m;
}

// ================================================================
extern "C" void kernel_launch(void* const* d_in, const int* in_sizes, int n_in,
                              void* d_out, int out_size, void* d_ws, size_t ws_size,
                              hipStream_t stream) {
  const float* xyz = (const float*)d_in[0];
  const float* pts = (const float*)d_in[1];
  const float* w0 = (const float*)d_in[2];  const float* b0 = (const float*)d_in[3];
  const float* g0 = (const float*)d_in[4];  const float* be0 = (const float*)d_in[5];
  const float* w1 = (const float*)d_in[6];  const float* b1 = (const float*)d_in[7];
  const float* g1 = (const float*)d_in[8];  const float* be1 = (const float*)d_in[9];
  const float* w2 = (const float*)d_in[10]; const float* b2 = (const float*)d_in[11];
  const float* g2 = (const float*)d_in[12]; const float* be2 = (const float*)d_in[13];

  char* ws = (char*)d_ws;
  float* sums  = (float*)(ws + O_SUMS);    // per layer: +l*256
  float* scale = (float*)(ws + O_SCALE);   // per layer: +l*256
  int*   fpsIdx = (int*)(ws + O_FPS);
  __hip_bfloat16* wb0 = (__hip_bfloat16*)(ws + O_WB0);
  __hip_bfloat16* wb1 = (__hip_bfloat16*)(ws + O_WB1);
  __hip_bfloat16* wb2 = (__hip_bfloat16*)(ws + O_WB2);
  __hip_bfloat16* x0  = (__hip_bfloat16*)(ws + O_X0);
  __hip_bfloat16* xb  = (__hip_bfloat16*)(ws + O_XB);
  __hip_bfloat16* y3  = (__hip_bfloat16*)(ws + O_Y3);

  float* newXyz = (float*)d_out;                               // (B, NPOINT, 3)
  float* newPts = (float*)d_out + BATCH * NPOINT * 3;          // (B, 128, NPOINT)

  prep_kernel<<<32, 256, 0, stream>>>(w0, w1, w2, wb0, wb1, wb2, sums);
  fps_kernel<<<BATCH, 1024, 0, stream>>>(xyz, fpsIdx, newXyz);
  ball_group_kernel<<<(BATCH * NPOINT) / 8, 256, 0, stream>>>(xyz, pts, newXyz, x0);

  const int gblocks = M_ROWS / 64;   // 16384
  // layer 0: 32 -> 64
  gemm_kernel<32, 64><<<gblocks, 128, 0, stream>>>(x0, wb0, b0, xb);
  stats_kernel<64><<<1024, 256, 0, stream>>>(xb, sums + 0 * 256);
  finalize_kernel<<<1, 128, 0, stream>>>(sums + 0 * 256, g0, be0, scale + 0 * 256, 64);
  bnrelu_kernel<64><<<2048, 256, 0, stream>>>(xb, scale + 0 * 256);
  // layer 1: 64 -> 64 (in place: block m touches only rows [64m,64m+64))
  gemm_kernel<64, 64><<<gblocks, 128, 0, stream>>>(xb, wb1, b1, xb);
  stats_kernel<64><<<1024, 256, 0, stream>>>(xb, sums + 1 * 256);
  finalize_kernel<<<1, 128, 0, stream>>>(sums + 1 * 256, g1, be1, scale + 1 * 256, 64);
  bnrelu_kernel<64><<<2048, 256, 0, stream>>>(xb, scale + 1 * 256);
  // layer 2: 64 -> 128
  gemm_kernel<64, 128><<<gblocks, 256, 0, stream>>>(xb, wb2, b2, y3);
  stats_kernel<128><<<1024, 256, 0, stream>>>(y3, sums + 2 * 256);
  finalize_kernel<<<1, 128, 0, stream>>>(sums + 2 * 256, g2, be2, scale + 2 * 256, 128);
  // fused BN+ReLU+max+transpose
  maxpool_kernel<<<(BATCH * NPOINT * 128) / 256, 256, 0, stream>>>(y3, scale + 2 * 256, newPts);
}
